// Cross_Attn_Image_to_Token_29807073034394
// MI455X (gfx1250) — matrix operand order repaired and checked by reading, not compiled, not run on hardware
//
#include <hip/hip_runtime.h>
#include <hip/hip_bf16.h>
#include <stdint.h>

// ---- problem constants (match reference) ----
#define B_      4
#define QLEN_   4096
#define KLEN_   1024
#define EMBED_  256
#define HEADS_  8
#define HD_     32
#define INTERNAL_ 256

typedef __attribute__((ext_vector_type(16))) __bf16 v16bf;
typedef __attribute__((ext_vector_type(8)))  float  v8f;

union FragBF {
  v16bf v;
  uint4 u[2];
};

__device__ __forceinline__ unsigned short f2bfbits(float f) {
  union { float f; uint32_t u; } c; c.f = f;
  uint32_t u = c.u + 0x7FFFu + ((c.u >> 16) & 1u); // round-to-nearest-even
  return (unsigned short)(u >> 16);
}

// ---------------- elementwise f32 -> bf16 convert ----------------
__global__ void __launch_bounds__(256) cvt_f32_bf16(const float* __restrict__ x,
                                                    unsigned short* __restrict__ y, int n) {
  int i = blockIdx.x * 256 + threadIdx.x;
  if (i < n) y[i] = f2bfbits(x[i]);
}

// ---------------- generic WMMA GEMM: Y = X[MxK] * W[NxK]^T + bias ----------------
// One wave computes a 16x64 strip (4 accumulators, A-fragment reused across 4 B tiles).
// MODE 0: qp -> split into q1/q2 head layout [b][h][q][d] bf16
// MODE 1: k/c -> head layout [b][h][key][d] bf16
// MODE 2: v   -> transposed head layout [b][h][d][key] bf16
// MODE 3: final -> f32 row-major [M][N]
template<int MODE>
__global__ void __launch_bounds__(128) gemm_wmma(
    const unsigned short* __restrict__ X,
    const unsigned short* __restrict__ W,
    const float* __restrict__ bias,
    void* __restrict__ out0, void* __restrict__ out1,
    int M, int N, int K)
{
  const int wave = blockIdx.x * (blockDim.x >> 5) + (threadIdx.x >> 5);
  const int lane = threadIdx.x & 31;
  const int half = lane >> 4;
  const int m    = lane & 15;
  const int ntiles = N >> 6;                 // strips of 64 columns
  const int nt = wave % ntiles;
  const int mt = wave / ntiles;
  if (mt >= (M >> 4)) return;
  const int m0 = mt << 4, n0 = nt << 6;

  v8f acc[4] = {{}, {}, {}, {}};
  const unsigned short* arow = X + (size_t)(m0 + m) * K;
  const unsigned short* brow0 = W + (size_t)(n0 + m) * K;

  for (int kc = 0; kc < K; kc += 32) {
    FragBF a;
    // A 16x32 bf16 layout: lanes 0-15 hold K 0..15, lanes 16-31 hold K 16..31
    a.u[0] = *(const uint4*)(arow + kc + half * 16);
    a.u[1] = *(const uint4*)(arow + kc + half * 16 + 8);
#pragma unroll
    for (int t = 0; t < 4; ++t) {
      // B 32x16 bf16 layout: lanes 0-15 hold K 0..15, lanes 16-31 hold K 16..31 (N = lane&15)
      const unsigned short* br = brow0 + (size_t)(t * 16) * K + kc + half * 16;
      FragBF b;
      b.u[0] = *(const uint4*)(br);
      b.u[1] = *(const uint4*)(br + 8);
      acc[t] = __builtin_amdgcn_wmma_f32_16x16x32_bf16(false, a.v, false, b.v,
                                                       (short)0, acc[t], false, false);
    }
  }

#pragma unroll
  for (int t = 0; t < 4; ++t) {
    const float bv = bias[n0 + t * 16 + m];
#pragma unroll
    for (int r = 0; r < 8; ++r) {
      const int row = m0 + r + (half << 3);   // D-tile: lane half selects rows 0-7 / 8-15
      const int col = n0 + t * 16 + m;
      const float val = acc[t][r] + bv;
      if (MODE == 0) {
        const int b = row >> 12, q = row & 4095;
        const int sel = col >> 8, jj = col & 255;
        const int h = jj >> 5, d = jj & 31;
        unsigned short* o = (unsigned short*)(sel ? out1 : out0);
        o[((((size_t)b * HEADS_ + h) * QLEN_) + q) * HD_ + d] = f2bfbits(val);
      } else if (MODE == 1) {
        const int b = row >> 10, key = row & 1023;
        const int h = col >> 5, d = col & 31;
        ((unsigned short*)out0)[((((size_t)b * HEADS_ + h) * KLEN_) + key) * HD_ + d] = f2bfbits(val);
      } else if (MODE == 2) {
        const int b = row >> 10, key = row & 1023;
        const int h = col >> 5, d = col & 31;
        ((unsigned short*)out0)[((((size_t)b * HEADS_ + h) * HD_) + d) * KLEN_ + key] = f2bfbits(val);
      } else {
        ((float*)out0)[(size_t)row * N + col] = val;
      }
    }
  }
}

// ---------------- online softmax update for one 16q x 32k score tile ----------------
__device__ __forceinline__ void online_update(
    const v8f& S0, const v8f& S1, float* mr, float* lr, v8f& oa, v8f& ob,
    unsigned short* P, int half, int m, float C)
{
#pragma unroll
  for (int r = 0; r < 8; ++r) {
    float t0 = S0[r] * C, t1 = S1[r] * C;        // scaled to log2 domain
    float a = fmaxf(t0, t1);
    a = fmaxf(a, __shfl_xor(a, 1));
    a = fmaxf(a, __shfl_xor(a, 2));
    a = fmaxf(a, __shfl_xor(a, 4));
    a = fmaxf(a, __shfl_xor(a, 8));              // row-max across 16-lane half
    const float mn = fmaxf(mr[r], a);
    const float sc = exp2f(mr[r] - mn);
    const float p0 = exp2f(t0 - mn);
    const float p1 = exp2f(t1 - mn);
    float rs = p0 + p1;
    rs += __shfl_xor(rs, 1);
    rs += __shfl_xor(rs, 2);
    rs += __shfl_xor(rs, 4);
    rs += __shfl_xor(rs, 8);                     // row-sum across half
    lr[r] = lr[r] * sc + rs;
    mr[r] = mn;
    oa[r] *= sc;
    ob[r] *= sc;
    const int row = r + (half << 3);
    P[row * 32 + m]      = f2bfbits(p0);
    P[row * 32 + 16 + m] = f2bfbits(p1);
  }
}

// ---- per-wave LDS region layout (bytes) ----
//   buf0: K @ 0 (2KB), C @ 2048 (2KB), V @ 4096 (2KB)
//   buf1: K @ 6144,    C @ 8192,       V @ 10240
//   P    : @ 12288 (1KB)   -> total 13312 B per wave
#define WREG_ 13312

// issue 12 async global->LDS b128 copies for one 32-key chunk of K, C, V^T
__device__ __forceinline__ void issue_async_chunk(
    uint32_t ldsbase, const unsigned short* khp, const unsigned short* chp,
    const unsigned short* vtp, int kc, int lane)
{
  // K and C chunks are 2048 contiguous bytes each ([key][dim] layout)
  const uint64_t gk = (uint64_t)(uintptr_t)(khp + (size_t)kc * HD_) + (uint32_t)(lane * 16);
  const uint64_t gc = (uint64_t)(uintptr_t)(chp + (size_t)kc * HD_) + (uint32_t)(lane * 16);
  const uint32_t lk = ldsbase + lane * 16;
  const uint32_t lc = ldsbase + 2048 + lane * 16;
#pragma unroll
  for (int j = 0; j < 4; ++j) {
    asm volatile("global_load_async_to_lds_b128 %0, %1, off"
                 :: "v"(lk + j * 512), "v"(gk + (uint64_t)(j * 512)) : "memory");
    asm volatile("global_load_async_to_lds_b128 %0, %1, off"
                 :: "v"(lc + j * 512), "v"(gc + (uint64_t)(j * 512)) : "memory");
  }
  // V^T chunk: 32 rows (dims) x 64B, row stride = KLEN*2 bytes
  const int l4 = lane & 3, l8 = lane >> 2;
#pragma unroll
  for (int j = 0; j < 4; ++j) {
    const int row = j * 8 + l8;
    const uint64_t gv = (uint64_t)(uintptr_t)(vtp + (size_t)row * KLEN_ + kc) + (uint32_t)(l4 * 16);
    const uint32_t lv = ldsbase + 4096 + row * 64 + l4 * 16;
    asm volatile("global_load_async_to_lds_b128 %0, %1, off"
                 :: "v"(lv), "v"(gv) : "memory");
  }
}

// ---------------- fused dual-stream flash attention ----------------
// One wave handles one (b, h, 16-row q-tile); 4 independent waves per block.
// K/C/V^T chunks are double-buffered in LDS via async loads (ASYNCcnt).
__global__ void __launch_bounds__(128) attn_flash_dual(
    const unsigned short* __restrict__ q1, const unsigned short* __restrict__ q2,
    const unsigned short* __restrict__ kh, const unsigned short* __restrict__ ch,
    const unsigned short* __restrict__ vhT,
    unsigned short* __restrict__ aout)
{
  __shared__ __align__(16) unsigned char smem[4][WREG_];
  const int wid  = threadIdx.x >> 5;
  const int wave = blockIdx.x * 4 + wid;
  const int lane = threadIdx.x & 31;
  const int half = lane >> 4;
  const int m    = lane & 15;

  const int qt = wave & 255;         // QLEN/16 = 256 tiles
  const int h  = (wave >> 8) & 7;
  const int b  = wave >> 11;
  const int q0 = qt << 4;
  const size_t bh = (size_t)b * HEADS_ + h;

  const unsigned short* q1p = q1 + (bh * QLEN_ + q0) * HD_;
  const unsigned short* q2p = q2 + (bh * QLEN_ + q0) * HD_;
  const unsigned short* khp = kh + bh * KLEN_ * HD_;
  const unsigned short* chp = ch + bh * KLEN_ * HD_;
  const unsigned short* vtp = vhT + bh * HD_ * KLEN_;

  const uint32_t wbase = (uint32_t)(uintptr_t)&smem[wid][0];
  unsigned short* P = (unsigned short*)&smem[wid][12288];

  // Q fragments: A 16x32 (full head dim = one WMMA K)
  FragBF A1, A2;
  {
    const unsigned short* r1 = q1p + m * HD_;
    const unsigned short* r2 = q2p + m * HD_;
    A1.u[0] = *(const uint4*)(r1 + half * 16);
    A1.u[1] = *(const uint4*)(r1 + half * 16 + 8);
    A2.u[0] = *(const uint4*)(r2 + half * 16);
    A2.u[1] = *(const uint4*)(r2 + half * 16 + 8);
  }

  const float C = 0.17677669529663687f * 1.4426950408889634f; // 1/sqrt(32) * log2(e)
  float m1[8], l1[8], m2[8], l2[8];
  v8f o1a = {}, o1b = {}, o2a = {}, o2b = {};
#pragma unroll
  for (int r = 0; r < 8; ++r) { m1[r] = m2[r] = -1.0e30f; l1[r] = l2[r] = 0.0f; }

  // prologue: async-stage chunk 0 into buffer 0
  issue_async_chunk(wbase, khp, chp, vtp, 0, lane);
  int buf = 0;

  for (int kc = 0; kc < KLEN_; kc += 32) {
    // overlap: kick off next chunk into the other buffer, then wait for current
    if (kc + 32 < KLEN_) {
      issue_async_chunk(wbase + (buf ^ 1) * 6144, khp, chp, vtp, kc + 32, lane);
      asm volatile("s_wait_asynccnt 0xc" ::: "memory");  // 12 outstanding = next chunk only
    } else {
      asm volatile("s_wait_asynccnt 0x0" ::: "memory");
    }

    const unsigned short* bK = (const unsigned short*)&smem[wid][buf * 6144];
    const unsigned short* bC = bK + 1024;   // +2048 bytes
    const unsigned short* bV = bK + 2048;   // +4096 bytes

    // K / C B-fragments from LDS (B = K^T: lane = key, K-dim = head dim)
    FragBF Bk0, Bk1, Bc0, Bc1;
    {
      const unsigned short* k0 = bK + m * 32 + half * 16;
      const unsigned short* k1 = bK + (16 + m) * 32 + half * 16;
      const unsigned short* c0 = bC + m * 32 + half * 16;
      const unsigned short* c1 = bC + (16 + m) * 32 + half * 16;
      Bk0.u[0] = *(const uint4*)(k0); Bk0.u[1] = *(const uint4*)(k0 + 8);
      Bk1.u[0] = *(const uint4*)(k1); Bk1.u[1] = *(const uint4*)(k1 + 8);
      Bc0.u[0] = *(const uint4*)(c0); Bc0.u[1] = *(const uint4*)(c0 + 8);
      Bc1.u[0] = *(const uint4*)(c1); Bc1.u[1] = *(const uint4*)(c1 + 8);
    }
    v8f z = {};
    v8f S10 = __builtin_amdgcn_wmma_f32_16x16x32_bf16(false, A1.v, false, Bk0.v, (short)0, z, false, false);
    v8f S11 = __builtin_amdgcn_wmma_f32_16x16x32_bf16(false, A1.v, false, Bk1.v, (short)0, z, false, false);
    v8f S20 = __builtin_amdgcn_wmma_f32_16x16x32_bf16(false, A2.v, false, Bc0.v, (short)0, z, false, false);
    v8f S21 = __builtin_amdgcn_wmma_f32_16x16x32_bf16(false, A2.v, false, Bc1.v, (short)0, z, false, false);

    // V B-fragments from LDS ([d][key] layout: lane = d, K-dim = keys)
    FragBF Bv0, Bv1;
    {
      const unsigned short* v0 = bV + m * 32 + half * 16;
      const unsigned short* v1 = bV + (16 + m) * 32 + half * 16;
      Bv0.u[0] = *(const uint4*)(v0); Bv0.u[1] = *(const uint4*)(v0 + 8);
      Bv1.u[0] = *(const uint4*)(v1); Bv1.u[1] = *(const uint4*)(v1 + 8);
    }

    // ---- stream 1: softmax update, stage P, P@V ----
    online_update(S10, S11, m1, l1, o1a, o1b, P, half, m, C);
    asm volatile("s_wait_dscnt 0" ::: "memory");
    FragBF Pf;
    {
      const unsigned short* pr = P + m * 32;
      Pf.u[0] = *(const uint4*)(pr + half * 16);
      Pf.u[1] = *(const uint4*)(pr + half * 16 + 8);
    }
    o1a = __builtin_amdgcn_wmma_f32_16x16x32_bf16(false, Pf.v, false, Bv0.v, (short)0, o1a, false, false);
    o1b = __builtin_amdgcn_wmma_f32_16x16x32_bf16(false, Pf.v, false, Bv1.v, (short)0, o1b, false, false);

    // ---- stream 2 ----
    online_update(S20, S21, m2, l2, o2a, o2b, P, half, m, C);
    asm volatile("s_wait_dscnt 0" ::: "memory");
    {
      const unsigned short* pr = P + m * 32;
      Pf.u[0] = *(const uint4*)(pr + half * 16);
      Pf.u[1] = *(const uint4*)(pr + half * 16 + 8);
    }
    o2a = __builtin_amdgcn_wmma_f32_16x16x32_bf16(false, Pf.v, false, Bv0.v, (short)0, o2a, false, false);
    o2b = __builtin_amdgcn_wmma_f32_16x16x32_bf16(false, Pf.v, false, Bv1.v, (short)0, o2b, false, false);

    buf ^= 1;
  }

  // epilogue: out = 0.5*(O1/l1 + O2/l2), store bf16 to [b][q][h*32+d]
#pragma unroll
  for (int r = 0; r < 8; ++r) {
    const int row = q0 + r + (half << 3);
    const size_t base = ((size_t)b * QLEN_ + row) * INTERNAL_ + h * HD_;
    const float i1 = 1.0f / l1[r];
    const float i2 = 1.0f / l2[r];
    aout[base + m]      = f2bfbits(0.5f * (o1a[r] * i1 + o2a[r] * i2));
    aout[base + 16 + m] = f2bfbits(0.5f * (o1b[r] * i1 + o2b[r] * i2));
  }
}

// ---------------- host orchestration ----------------
extern "C" void kernel_launch(void* const* d_in, const int* in_sizes, int n_in,
                              void* d_out, int out_size, void* d_ws, size_t ws_size,
                              hipStream_t stream) {
  const float* q    = (const float*)d_in[0];
  const float* k    = (const float*)d_in[1];
  const float* v    = (const float*)d_in[2];
  const float* cond = (const float*)d_in[3];
  const float* Wq2  = (const float*)d_in[4];
  const float* bq2  = (const float*)d_in[5];
  const float* Wk   = (const float*)d_in[6];
  const float* bk   = (const float*)d_in[7];
  const float* Wc   = (const float*)d_in[8];
  const float* bc   = (const float*)d_in[9];
  const float* Wv   = (const float*)d_in[10];
  const float* bv   = (const float*)d_in[11];
  const float* Wo   = (const float*)d_in[12];
  const float* bo   = (const float*)d_in[13];

  const size_t NQ  = (size_t)B_ * QLEN_ * EMBED_;        // 4,194,304
  const size_t NK  = (size_t)B_ * KLEN_ * EMBED_;        // 1,048,576
  const size_t NH  = (size_t)B_ * HEADS_ * KLEN_ * HD_;  // 1,048,576
  const size_t NQH = (size_t)B_ * HEADS_ * QLEN_ * HD_;  // 4,194,304

  size_t off = 0;
  auto take = [&](size_t elems) {
    size_t o = off; off += ((elems * 2) + 255) & ~(size_t)255; return o;
  };
  char* ws = (char*)d_ws;
  unsigned short* qbf   = (unsigned short*)(ws + take(NQ));
  unsigned short* kbf   = (unsigned short*)(ws + take(NK));
  unsigned short* vbf   = (unsigned short*)(ws + take(NK));
  unsigned short* cbf   = (unsigned short*)(ws + take(NK));
  unsigned short* wq2b  = (unsigned short*)(ws + take(512 * 256));
  unsigned short* wkb   = (unsigned short*)(ws + take(256 * 256));
  unsigned short* wcb   = (unsigned short*)(ws + take(256 * 256));
  unsigned short* wvb   = (unsigned short*)(ws + take(256 * 256));
  unsigned short* wob   = (unsigned short*)(ws + take(256 * 256));
  unsigned short* q1w   = (unsigned short*)(ws + take(NQH));
  unsigned short* q2w   = (unsigned short*)(ws + take(NQH));
  unsigned short* khw   = (unsigned short*)(ws + take(NH));
  unsigned short* chw   = (unsigned short*)(ws + take(NH));
  unsigned short* vtw   = (unsigned short*)(ws + take(NH));
  unsigned short* aoutw = (unsigned short*)(ws + take(NQ));

  auto cvt = [&](const float* src, unsigned short* dst, size_t n) {
    cvt_f32_bf16<<<(unsigned)((n + 255) / 256), 256, 0, stream>>>(src, dst, (int)n);
  };
  cvt(q, qbf, NQ);
  cvt(k, kbf, NK);
  cvt(v, vbf, NK);
  cvt(cond, cbf, NK);
  cvt(Wq2, wq2b, 512 * 256);
  cvt(Wk, wkb, 256 * 256);
  cvt(Wc, wcb, 256 * 256);
  cvt(Wv, wvb, 256 * 256);
  cvt(Wo, wob, 256 * 256);

  // qp = q @ Wq2^T + bq2  -> q1/q2 head layout   (M=16384, N=512, K=256)
  {
    int M = B_ * QLEN_, N = 2 * INTERNAL_, K = EMBED_;
    int waves = (M / 16) * (N / 64);
    gemm_wmma<0><<<waves / 4, 128, 0, stream>>>(qbf, wq2b, bq2, q1w, q2w, M, N, K);
  }
  // kh, ch (head layout), vhT (transposed head layout)   (M=4096, N=256, K=256)
  {
    int M = B_ * KLEN_, N = INTERNAL_, K = EMBED_;
    int waves = (M / 16) * (N / 64);
    gemm_wmma<1><<<waves / 4, 128, 0, stream>>>(kbf, wkb, bk, khw, nullptr, M, N, K);
    gemm_wmma<1><<<waves / 4, 128, 0, stream>>>(cbf, wcb, bc, chw, nullptr, M, N, K);
    gemm_wmma<2><<<waves / 4, 128, 0, stream>>>(vbf, wvb, bv, vtw, nullptr, M, N, K);
  }
  // fused dual-stream flash attention: 8192 waves, 4 per block
  attn_flash_dual<<<2048, 128, 0, stream>>>(q1w, q2w, khw, chw, vtw, aoutw);

  // final projection: d_out = attn_out @ Wo^T + bo  (f32 out)   (M=16384, N=256, K=256)
  {
    int M = B_ * QLEN_, N = EMBED_, K = INTERNAL_;
    int waves = (M / 16) * (N / 64);
    gemm_wmma<3><<<waves / 4, 128, 0, stream>>>(aoutw, wob, bo, d_out, nullptr, M, N, K);
  }
}
